// MultiHeadAttention_11630771437703
// MI455X (gfx1250) — matrix operand order repaired and checked by reading, not compiled, not run on hardware
//
#include <hip/hip_runtime.h>
#include <hip/hip_bf16.h>
#include <hip/hip_fp16.h>

typedef __attribute__((ext_vector_type(16))) _Float16 v16h;
typedef __attribute__((ext_vector_type(8)))  _Float16 v8h;
typedef __attribute__((ext_vector_type(8)))  float    v8f;
typedef __attribute__((ext_vector_type(4)))  float    v4f;

#define DEVINL __device__ __forceinline__

constexpr int B_ = 4, L_ = 2048, D_ = 1024, H_ = 16, DK_ = 64;
constexpr int MTOT = B_ * L_;            // 8192 rows
constexpr float LNEPS = 1e-5f;
// 1/sqrt(DK) * log2(e): scores are produced pre-scaled for base-2 softmax
constexpr float QSCALE = 0.125f * 1.44269504088896340736f;

DEVINL float fast_max(float a, float b) { return a >= b ? a : b; }

// ---------------------------------------------------------------------------
// WMMA helpers: wave32 16x16x32 f16->f32 fragment layouts (CDNA5 ISA)
// A (16x32, 16-bit): lane = h*16+m; vgprs 0..3 hold k = h*8 + {0..7},
//                    vgprs 4..7 hold k = 16 + h*8 + {0..7}.
// B (32x16, 16-bit): lane = h*16+n; same k order as A (h*8+{0..7}, 16+h*8+{0..7}).
// C/D (16x16 f32):   lane = h*16+n; vgpr r holds row m = h*8 + r.
// ---------------------------------------------------------------------------
DEVINL v8f wmma_f16(v16h a, v16h b, v8f c) {
  return __builtin_amdgcn_wmma_f32_16x16x32_f16(false, a, false, b,
                                                (short)0, c, false, false);
}

DEVINL v16h frag_a_f16(const _Float16* p, int ld) {
  const int lane = threadIdx.x & 31, h = lane >> 4, m = lane & 15;
  const _Float16* r = p + (size_t)m * ld + h * 8;
  v8h lo = *(const v8h*)r;
  v8h hi = *(const v8h*)(r + 16);
  v16h a;
#pragma unroll
  for (int i = 0; i < 8; ++i) { a[i] = lo[i]; a[i + 8] = hi[i]; }
  return a;
}

DEVINL v16h frag_a_f32(const float* p, int ld) {
  const int lane = threadIdx.x & 31, h = lane >> 4, m = lane & 15;
  const float* r = p + (size_t)m * ld + h * 8;
  v4f x0 = *(const v4f*)(r);
  v4f x1 = *(const v4f*)(r + 4);
  v4f y0 = *(const v4f*)(r + 16);
  v4f y1 = *(const v4f*)(r + 20);
  v16h a;
#pragma unroll
  for (int i = 0; i < 4; ++i) {
    a[i]      = (_Float16)x0[i];
    a[i + 4]  = (_Float16)x1[i];
    a[i + 8]  = (_Float16)y0[i];
    a[i + 12] = (_Float16)y1[i];
  }
  return a;
}

// B fragment from a row-major [n][k] source (row n contiguous in k):
// covers B[k][n] = SRC[n][k] for k = 0..31, n = 0..15.
DEVINL v16h frag_b_f16(const _Float16* p, int ld) {
  const int lane = threadIdx.x & 31, h = lane >> 4, n = lane & 15;
  const _Float16* r = p + (size_t)n * ld + h * 8;
  v8h lo = *(const v8h*)r;
  v8h hi = *(const v8h*)(r + 16);
  v16h b;
#pragma unroll
  for (int i = 0; i < 8; ++i) { b[i] = lo[i]; b[i + 8] = hi[i]; }
  return b;
}

// ---------------------------------------------------------------------------
// f32 -> f16 weight conversion
// ---------------------------------------------------------------------------
__global__ void cvt_f32_f16(const float* __restrict__ src,
                            _Float16* __restrict__ dst, int n) {
  int i = blockIdx.x * 256 + threadIdx.x;
  if (i < n) dst[i] = (_Float16)src[i];
}

// ---------------------------------------------------------------------------
// GEMM: Y[m][n] = (sum_k X[m][k] * W[n][k] + bias[n]) * scale
// X: [8192][1024] (f32 input or f16 ws), W: f16 [1024][1024] torch [out,in].
// 8 waves/block; wave = 32x32 output (2x2 WMMA); block tile 64(M) x 128(N).
// VTOUT: store transposed per-head f16 V^T[b][n][l]  (n = h*64+d).
// ---------------------------------------------------------------------------
template <bool A16, bool VTOUT, bool F32OUT>
__global__ __launch_bounds__(256) void gemm_kernel(
    const void* __restrict__ Xv, const _Float16* __restrict__ Wt,
    const float* __restrict__ bias, void* __restrict__ Yv, float scale) {
  const int lane = threadIdx.x & 31, w = threadIdx.x >> 5;
  const int wm = w & 1, wn = w >> 1;
  const int mbase = blockIdx.y * 64 + wm * 32;
  const int nbase = blockIdx.x * 128 + wn * 32;
  const float*    Xf = (const float*)Xv;
  const _Float16* Xh = (const _Float16*)Xv;

  v8f acc[2][2] = {};
  for (int kt = 0; kt < D_; kt += 32) {
    v16h a0, a1;
    if constexpr (A16) {
      a0 = frag_a_f16(Xh + (size_t)mbase * D_ + kt, D_);
      a1 = frag_a_f16(Xh + (size_t)(mbase + 16) * D_ + kt, D_);
    } else {
      a0 = frag_a_f32(Xf + (size_t)mbase * D_ + kt, D_);
      a1 = frag_a_f32(Xf + (size_t)(mbase + 16) * D_ + kt, D_);
    }
    v16h b0 = frag_b_f16(Wt + (size_t)nbase * D_ + kt, D_);
    v16h b1 = frag_b_f16(Wt + (size_t)(nbase + 16) * D_ + kt, D_);
    if (kt + 32 < D_) {   // next-tile prefetch -> global_prefetch_b8
      __builtin_prefetch(Wt + (size_t)nbase * D_ + kt + 32, 0, 1);
      __builtin_prefetch(Wt + (size_t)(nbase + 16) * D_ + kt + 32, 0, 1);
    }
    acc[0][0] = wmma_f16(a0, b0, acc[0][0]);
    acc[0][1] = wmma_f16(a0, b1, acc[0][1]);
    acc[1][0] = wmma_f16(a1, b0, acc[1][0]);
    acc[1][1] = wmma_f16(a1, b1, acc[1][1]);
  }

  const int hc = lane >> 4, nn = lane & 15;
#pragma unroll
  for (int i = 0; i < 2; ++i) {
#pragma unroll
    for (int j = 0; j < 2; ++j) {
      const int ng = nbase + j * 16 + nn;
      const float bv = bias[ng];
#pragma unroll
      for (int r = 0; r < 8; ++r) {
        const int mg = mbase + i * 16 + hc * 8 + r;
        const float val = (acc[i][j][r] + bv) * scale;
        if constexpr (F32OUT) {
          ((float*)Yv)[(size_t)mg * D_ + ng] = val;
        } else if constexpr (VTOUT) {
          const int bb = mg >> 11, l = mg & (L_ - 1);
          ((_Float16*)Yv)[((size_t)bb * D_ + ng) * L_ + l] = (_Float16)val;
        } else {
          ((_Float16*)Yv)[(size_t)mg * D_ + ng] = (_Float16)val;
        }
      }
    }
  }
}

// ---------------------------------------------------------------------------
// Attention. grid (L/128, H, B); block = 256 threads = 8 waves = 128 q-rows.
// Pass 1: K head streamed through LDS in 128-key chunks with CDNA5 async
//   copies (global_load_async_to_lds_b128 / ASYNCcnt); softmax max/sum kept
//   as lane-local online stats in base-2 (scores pre-scaled by log2(e)),
//   one cross-lane combine at the end.
// Pass 2: recompute S, write normalized P to d_out once, P->A-fragment via a
//   1KB wave-private LDS bounce (s_wait_dscnt), accumulate P*V from V^T.
// ---------------------------------------------------------------------------
constexpr int KCH = 128;  // keys staged per chunk (16KB f16)

__global__ __launch_bounds__(256) void attn_kernel(
    const _Float16* __restrict__ QH, const _Float16* __restrict__ KH,
    const _Float16* __restrict__ VT, float* __restrict__ attn_out,
    _Float16* __restrict__ O16) {
  __shared__ alignas(16) _Float16 Ksh[KCH][DK_];      // 16 KB
  __shared__ alignas(16) _Float16 Plds[8][16 * 32];   // 8 KB
  const int tid = threadIdx.x;
  const int lane = tid & 31, w = tid >> 5;
  const int hc = lane >> 4, nn = lane & 15;
  const int head = blockIdx.y, b = blockIdx.z;
  const int mwave = blockIdx.x * 128 + w * 16;
  const size_t qrow0 = (size_t)(b * L_ + mwave);

  // Q fragments (pre-scaled by log2(e)/sqrt(DK) at projection time)
  const v16h Qa0 = frag_a_f16(QH + qrow0 * D_ + head * DK_, D_);
  const v16h Qa1 = frag_a_f16(QH + qrow0 * D_ + head * DK_ + 32, D_);

  // ---- pass 1: lane-local online row max / sum over all 2048 keys ----
  float lm[8], ls[8];
#pragma unroll
  for (int r = 0; r < 8; ++r) { lm[r] = -3.0e38f; ls[r] = 0.f; }

  const unsigned ksh_base = (unsigned)(size_t)&Ksh[0][0];
  for (int kc = 0; kc < L_; kc += KCH) {
    __syncthreads();  // all waves done reading previous chunk
    // cooperative async copy: KCH x 64 halfs = 16KB = 1024 x 16B chunks
#pragma unroll
    for (int i = 0; i < 4; ++i) {
      const int chunk = tid + i * 256;
      const int row = chunk >> 3, cb = (chunk & 7) * 16;  // bytes within row
      const unsigned ldsa = ksh_base + (unsigned)chunk * 16u;
      const void* gp = (const char*)(KH +
          ((size_t)(b * L_ + kc + row)) * D_ + head * DK_) + cb;
      asm volatile("global_load_async_to_lds_b128 %0, %1, off"
                   :: "v"(ldsa), "v"((unsigned long long)(size_t)gp)
                   : "memory");
    }
    asm volatile("s_wait_asynccnt 0x0" ::: "memory");
    __syncthreads();  // chunk visible to every wave

#pragma unroll
    for (int t = 0; t < KCH / 16; ++t) {
      const _Float16* kp = &Ksh[t * 16][0];
      v8f s = {};
      s = wmma_f16(Qa0, frag_b_f16(kp, DK_), s);
      s = wmma_f16(Qa1, frag_b_f16(kp + 32, DK_), s);
#pragma unroll
      for (int r = 0; r < 8; ++r) {
        const float sv = s[r];
        const float mn = fast_max(lm[r], sv);
        ls[r] = ls[r] * exp2f(lm[r] - mn) + exp2f(sv - mn);
        lm[r] = mn;
      }
    }
  }
  // cross-lane combine (16-lane halves share a row)
  float rowm[8], rinv[8];
#pragma unroll
  for (int r = 0; r < 8; ++r) {
    float M = lm[r];
#pragma unroll
    for (int m = 1; m < 16; m <<= 1) M = fast_max(M, __shfl_xor(M, m));
    float S = ls[r] * exp2f(lm[r] - M);
#pragma unroll
    for (int m = 1; m < 16; m <<= 1) S += __shfl_xor(S, m);
    rowm[r] = M;
    rinv[r] = 1.0f / S;
  }

  // ---- pass 2: recompute S, write normalized P once, accumulate P*V ----
  v8f oacc[4] = {};
  float* Pout = attn_out + ((size_t)(head * B_ + b) * L_ + mwave) * (size_t)L_;
  for (int kb = 0; kb < L_; kb += 32) {
    const _Float16* kp = KH + ((size_t)(b * L_ + kb)) * D_ + head * DK_;
    v8f s0 = {}, s1 = {};
    s0 = wmma_f16(Qa0, frag_b_f16(kp, D_), s0);
    s0 = wmma_f16(Qa1, frag_b_f16(kp + 32, D_), s0);
    s1 = wmma_f16(Qa0, frag_b_f16(kp + (size_t)16 * D_, D_), s1);
    s1 = wmma_f16(Qa1, frag_b_f16(kp + (size_t)16 * D_ + 32, D_), s1);
#pragma unroll
    for (int r = 0; r < 8; ++r) {
      const int row = hc * 8 + r;
      const float p0 = exp2f(s0[r] - rowm[r]) * rinv[r];
      const float p1 = exp2f(s1[r] - rowm[r]) * rinv[r];
      Pout[(size_t)row * L_ + kb + nn]      = p0;
      Pout[(size_t)row * L_ + kb + 16 + nn] = p1;
      Plds[w][row * 32 + nn]      = (_Float16)p0;
      Plds[w][row * 32 + 16 + nn] = (_Float16)p1;
    }
    // wave-private LDS region: stores are in-order with loads, drain DScnt
    asm volatile("s_wait_dscnt 0x0" ::: "memory");
    const v16h pa = frag_a_f16(&Plds[w][0], 32);
#pragma unroll
    for (int nt = 0; nt < 4; ++nt) {
      const _Float16* vp =
          VT + ((size_t)(b * D_ + head * DK_ + nt * 16)) * L_ + kb;
      oacc[nt] = wmma_f16(pa, frag_b_f16(vp, L_), oacc[nt]);
    }
  }

#pragma unroll
  for (int nt = 0; nt < 4; ++nt) {
#pragma unroll
    for (int r = 0; r < 8; ++r) {
      const int mg = b * L_ + mwave + hc * 8 + r;
      O16[(size_t)mg * D_ + head * DK_ + nt * 16 + nn] = (_Float16)oacc[nt][r];
    }
  }
}

// ---------------------------------------------------------------------------
// In-place LayerNorm + residual on d_out rows. 1 block per row.
// ---------------------------------------------------------------------------
__global__ __launch_bounds__(256) void ln_kernel(
    float* __restrict__ Y, const float* __restrict__ resid,
    const float* __restrict__ g, const float* __restrict__ bb) {
  __shared__ float redA[8], redB[8];
  const int row = blockIdx.x, tid = threadIdx.x;
  const int lane = tid & 31, w = tid >> 5;
  const size_t base = (size_t)row * D_;
  float vals[4], s = 0.f, q2 = 0.f;
#pragma unroll
  for (int i = 0; i < 4; ++i) {
    float v = Y[base + tid + i * 256];
    vals[i] = v; s += v; q2 += v * v;
  }
#pragma unroll
  for (int m = 1; m < 32; m <<= 1) { s += __shfl_xor(s, m); q2 += __shfl_xor(q2, m); }
  if (lane == 0) { redA[w] = s; redB[w] = q2; }
  __syncthreads();
  if (tid == 0) {
    float a = 0.f, c = 0.f;
    for (int i = 0; i < 8; ++i) { a += redA[i]; c += redB[i]; }
    redA[0] = a; redB[0] = c;
  }
  __syncthreads();
  const float mean = redA[0] * (1.0f / D_);
  const float var  = redB[0] * (1.0f / D_) - mean * mean;
  const float rstd = rsqrtf(var + LNEPS);
#pragma unroll
  for (int i = 0; i < 4; ++i) {
    const int c = tid + i * 256;
    Y[base + c] = (vals[i] - mean) * rstd * g[c] + bb[c] + resid[base + c];
  }
}

// ---------------------------------------------------------------------------
extern "C" void kernel_launch(void* const* d_in, const int* in_sizes, int n_in,
                              void* d_out, int out_size, void* d_ws,
                              size_t ws_size, hipStream_t stream) {
  (void)in_sizes; (void)n_in; (void)out_size; (void)ws_size;
  const float* q  = (const float*)d_in[0];
  const float* k  = (const float*)d_in[1];
  const float* v  = (const float*)d_in[2];
  const float* wq = (const float*)d_in[3];
  const float* bq = (const float*)d_in[4];
  const float* wk = (const float*)d_in[5];
  const float* bk = (const float*)d_in[6];
  const float* wv = (const float*)d_in[7];
  const float* bv = (const float*)d_in[8];
  const float* wo = (const float*)d_in[9];
  const float* bo = (const float*)d_in[10];
  const float* lg = (const float*)d_in[11];
  const float* lb = (const float*)d_in[12];

  float* out  = (float*)d_out;                       // [B,L,D] f32
  float* attn = out + (size_t)B_ * L_ * D_;          // [H*B, L, L] f32

  // workspace layout (halfs)
  constexpr size_t WSZ = (size_t)D_ * D_;            // 1,048,576
  constexpr size_t XSZ = (size_t)MTOT * D_;          // 8,388,608
  _Float16* WQ16 = (_Float16*)d_ws;
  _Float16* WK16 = WQ16 + WSZ;
  _Float16* WV16 = WK16 + WSZ;
  _Float16* WO16 = WV16 + WSZ;
  _Float16* QH   = WO16 + WSZ;   // [8192][1024], pre-scaled log2(e)/sqrt(DK)
  _Float16* KH   = QH + XSZ;     // [8192][1024]
  _Float16* VT   = KH + XSZ;     // [B][1024][2048] transposed heads
  _Float16* O16  = VT + XSZ;     // [8192][1024]

  const int wn = (int)WSZ;
  cvt_f32_f16<<<(wn + 255) / 256, 256, 0, stream>>>(wq, WQ16, wn);
  cvt_f32_f16<<<(wn + 255) / 256, 256, 0, stream>>>(wk, WK16, wn);
  cvt_f32_f16<<<(wn + 255) / 256, 256, 0, stream>>>(wv, WV16, wn);
  cvt_f32_f16<<<(wn + 255) / 256, 256, 0, stream>>>(wo, WO16, wn);

  const dim3 gg(D_ / 128, MTOT / 64);  // (8, 128)
  gemm_kernel<false, false, false><<<gg, 256, 0, stream>>>(q, WQ16, bq, QH, QSCALE);
  gemm_kernel<false, false, false><<<gg, 256, 0, stream>>>(k, WK16, bk, KH, 1.0f);
  gemm_kernel<false, true,  false><<<gg, 256, 0, stream>>>(v, WV16, bv, VT, 1.0f);

  attn_kernel<<<dim3(L_ / 128, H_, B_), 256, 0, stream>>>(QH, KH, VT, attn, O16);

  gemm_kernel<true, false, true><<<gg, 256, 0, stream>>>(O16, WO16, bo, out, 1.0f);

  ln_kernel<<<MTOT, 256, 0, stream>>>(out, q, lg, lb);
}
